// MultiHeadAttentionRoPE_18949395710534
// MI455X (gfx1250) — compile-verified, matrix operand order checked by reading, not run on hardware
//
#include <hip/hip_runtime.h>

// ---------------------------------------------------------------------------
// MultiHeadAttention + RoPE for MI455X (gfx1250, wave32, WMMA bf16)
// B=2, T=2048, C=1024, H=16, D=64
// ---------------------------------------------------------------------------

typedef __attribute__((ext_vector_type(16))) __bf16 v16bf;
typedef __attribute__((ext_vector_type(8)))  float  v8f;

union ABFrag { v16bf v; uint4 q[2]; };
union CFrag  { v8f   v; float f[8]; };

__device__ __forceinline__ unsigned short f2bf(float x) {
    unsigned int u = __float_as_uint(x);
    u += 0x7fffu + ((u >> 16) & 1u);          // round-to-nearest-even
    return (unsigned short)(u >> 16);
}

// Load a 16x32 bf16 A/B fragment from LDS (row-major tile, `stride` in
// ushorts). Per ISA 7.12.2: lanes 0-15 hold M=lane, K=[0..7]+[16..23];
// lanes 16-31 hold M=lane-16, K=[8..15]+[24..31]. Two 16B LDS loads per lane.
__device__ __forceinline__ ABFrag load_frag16(const unsigned short* lds,
                                              int row0, int stride, int k0,
                                              int lane) {
    const int hl = lane >> 4;
    const int m  = row0 + (lane & 15);
    const int kb = k0 + 8 * hl;
    ABFrag f;
    f.q[0] = *(const uint4*)(lds + m * stride + kb);
    f.q[1] = *(const uint4*)(lds + m * stride + kb + 16);
    return f;
}

// ---------------------------------------------------------------------------
// 128x128x32 WMMA GEMM core, 512 threads = 16 waves arranged 4(M) x 4(N).
// Each wave owns a 32x32 sub-tile = 4 C-frags (32 VGPRs) so accumulators
// stay register-resident under the observed ~96-VGPR allocation budget.
// Threads 0-255 stream the A tile to LDS, threads 256-511 the B^T tile.
// ---------------------------------------------------------------------------
__device__ __forceinline__ void gemm_tile_128x128(
    const unsigned short* __restrict__ Ag,   // [M][K] bf16 row-major
    const unsigned short* __restrict__ Bg,   // [N][K] bf16 row-major (B^T)
    int K, int blockM, int blockN,
    CFrag (&c)[2][2], unsigned short* sA, unsigned short* sB) {

    const int tid   = threadIdx.x;
    const int lane  = tid & 31;
    const int wave  = tid >> 5;
    const int waveM = wave >> 2;          // 0..3
    const int waveN = wave & 3;           // 0..3

    for (int mi = 0; mi < 2; ++mi)
        for (int ni = 0; ni < 2; ++ni)
            for (int r = 0; r < 8; ++r) c[mi][ni].f[r] = 0.0f;

    const int isB  = tid >> 8;                 // 0: A-loader, 1: B-loader
    const int lrow = (tid & 255) >> 1;         // 0..127
    const int lcol = (tid & 1) * 16;           // 0 or 16 (ushorts)
    const unsigned short* gsrc = isB ? Bg : Ag;
    const int gbase = isB ? blockN : blockM;
    unsigned short* ldst = isB ? sB : sA;

    for (int k0 = 0; k0 < K; k0 += 32) {
        const uint4* g = (const uint4*)(gsrc + (size_t)(gbase + lrow) * K + k0 + lcol);
        uint4 t0 = g[0], t1 = g[1];
        if (k0 + 32 < K)   // global_prefetch_b8 next K tile
            __builtin_prefetch(gsrc + (size_t)(gbase + lrow) * K + k0 + 32 + lcol, 0, 1);
        __syncthreads();   // previous iteration's frag reads done
        *(uint4*)(ldst + lrow * 32 + lcol)     = t0;
        *(uint4*)(ldst + lrow * 32 + lcol + 8) = t1;
        __syncthreads();

        ABFrag b0 = load_frag16(sB, waveN * 32,      32, 0, lane);
        ABFrag b1 = load_frag16(sB, waveN * 32 + 16, 32, 0, lane);
        for (int mi = 0; mi < 2; ++mi) {
            ABFrag a = load_frag16(sA, waveM * 32 + mi * 16, 32, 0, lane);
            c[mi][0].v = __builtin_amdgcn_wmma_f32_16x16x32_bf16(
                false, a.v, false, b0.v, (short)0, c[mi][0].v, false, false);
            c[mi][1].v = __builtin_amdgcn_wmma_f32_16x16x32_bf16(
                false, a.v, false, b1.v, (short)0, c[mi][1].v, false, false);
        }
    }
}

// ---------------------------------------------------------------------------
// Converters
// ---------------------------------------------------------------------------
__global__ __launch_bounds__(256) void convert_f32_bf16_kernel(
    const float* __restrict__ in, unsigned short* __restrict__ out, int n4) {
    int i = blockIdx.x * 256 + threadIdx.x;
    if (i < n4) {
        float4 v = ((const float4*)in)[i];
        out[4 * i + 0] = f2bf(v.x);
        out[4 * i + 1] = f2bf(v.y);
        out[4 * i + 2] = f2bf(v.z);
        out[4 * i + 3] = f2bf(v.w);
    }
}

// in [R][C] f32  ->  out [C][R] bf16 (32x32 LDS tile transpose)
__global__ __launch_bounds__(256) void transpose_convert_kernel(
    const float* __restrict__ in, unsigned short* __restrict__ out,
    int R, int C) {
    __shared__ float tile[32][33];
    const int bx = blockIdx.x * 32;          // C offset
    const int by = blockIdx.y * 32;          // R offset
    const int tx  = threadIdx.x & 31;
    const int ty4 = (threadIdx.x >> 5) * 4;  // 8 waves * 4 rows
    for (int i = 0; i < 4; ++i)
        tile[ty4 + i][tx] = in[(size_t)(by + ty4 + i) * C + bx + tx];
    __syncthreads();
    for (int i = 0; i < 4; ++i)
        out[(size_t)(bx + ty4 + i) * R + by + tx] = f2bf(tile[tx][ty4 + i]);
}

// ---------------------------------------------------------------------------
// Kernel 1: QKV projection GEMM + fused RoPE/split epilogue
// ---------------------------------------------------------------------------
__global__ __launch_bounds__(512)
__attribute__((amdgpu_waves_per_eu(1)))
void gemm_qkv_rope_kernel(
    const unsigned short* __restrict__ xb,
    const unsigned short* __restrict__ wT,
    unsigned short* __restrict__ qp,
    unsigned short* __restrict__ kp,
    unsigned short* __restrict__ vtp) {
    __shared__ unsigned short sA[128 * 32];
    __shared__ unsigned short sB[128 * 32];
    CFrag c[2][2];
    const int blockM = blockIdx.y * 128, blockN = blockIdx.x * 128;
    gemm_tile_128x128(xb, wT, 1024, blockM, blockN, c, sA, sB);

    const int lane  = threadIdx.x & 31;
    const int wave  = threadIdx.x >> 5;
    const int waveM = wave >> 2, waveN = wave & 3;
    const int hl    = lane >> 4, l15 = lane & 15;

    for (int mi = 0; mi < 2; ++mi) {
        const int rowBase = blockM + waveM * 32 + mi * 16 + 8 * hl;
        for (int ni = 0; ni < 2; ++ni) {
            const int col   = blockN + waveN * 32 + ni * 16 + l15;
            const int which = col >> 10;          // 0=q 1=k 2=v (wave-uniform)
            const int cw    = col & 1023;
            const int h     = cw >> 6;
            const int d     = cw & 63;
            for (int r = 0; r < 8; ++r) {
                const int row = rowBase + r;
                const int b   = row >> 11;         // /2048
                const int t   = row & 2047;
                const int bh  = b * 16 + h;
                float val = c[mi][ni].f[r];
                if (which == 2) {
                    vtp[((size_t)(bh * 64 + d)) * 2048 + t] = f2bf(val);
                } else {
                    // RoPE: pair element lives in adjacent lane (d ^ 1)
                    float partner = __shfl_xor(val, 1, 32);
                    const int j = d >> 1;
                    // inv_freq = 10000^(-j/32) = exp(-j*ln(10000)/32)
                    float ang = (float)t *
                        __expf(-(float)j * (9.210340371976184f / 32.0f));
                    float sn, cs;
                    __sincosf(ang, &sn, &cs);
                    float o = (d & 1) ? (val * cs + partner * sn)
                                      : (val * cs - partner * sn);
                    unsigned short* dst = (which == 0) ? qp : kp;
                    dst[(((size_t)bh * 2048 + t)) * 64 + d] = f2bf(o);
                }
            }
        }
    }
}

// ---------------------------------------------------------------------------
// Kernel 2: flash attention. Block = 128 thr (4 waves) = 64 queries.
// Grid (32 q-tiles, 16 heads, 2 batch). S is staged through per-wave LDS
// (scale + causal mask applied at store) so only one S-frag is live at a
// time; Q-frags reloaded from LDS per KV tile. Keeps peak < ~90 VGPRs so
// the 4 O accumulator frags never spill.
// ---------------------------------------------------------------------------
__global__ __launch_bounds__(128)
__attribute__((amdgpu_waves_per_eu(1)))
void flash_attn_kernel(
    const unsigned short* __restrict__ qg,
    const unsigned short* __restrict__ kg,
    const unsigned short* __restrict__ vtg,
    unsigned short* __restrict__ attn) {
    __shared__ unsigned short sQ[64 * 64];
    __shared__ unsigned short sK[64 * 64];
    __shared__ unsigned short sVT[64 * 64];
    __shared__ unsigned short sP[4 * 16 * 64];   // per-wave P tile (bf16)
    __shared__ float          sS[4 * 16 * 64];   // per-wave S tile (f32)

    const int qt = blockIdx.x, h = blockIdx.y, b = blockIdx.z;
    const int bh = b * 16 + h;
    const int tid = threadIdx.x, lane = tid & 31, wave = tid >> 5;
    const int hl = lane >> 4, l15 = lane & 15;
    const int q0 = qt * 64;

    {   // load Q tile [64 x 64]
        const int row = tid >> 1, hc = (tid & 1) * 32;
        const uint4* g = (const uint4*)(qg + ((size_t)bh * 2048 + q0 + row) * 64 + hc);
        uint4* dd = (uint4*)(&sQ[row * 64 + hc]);
        dd[0] = g[0]; dd[1] = g[1]; dd[2] = g[2]; dd[3] = g[3];
    }
    __syncthreads();

    float mrow[8], lrow[8];
    CFrag o[4];
    for (int r = 0; r < 8; ++r) { mrow[r] = -1e30f; lrow[r] = 0.0f; }
    for (int nf = 0; nf < 4; ++nf)
        for (int r = 0; r < 8; ++r) o[nf].f[r] = 0.0f;
    unsigned short* myP = sP + wave * 1024;
    float*          myS = sS + wave * 1024;

    for (int j = 0; j <= qt; ++j) {
        const int kt0 = j * 64;
        {   // load K tile [64 keys x 64 d] and V^T tile [64 d x 64 keys]
            const int row = tid >> 1, hc = (tid & 1) * 32;
            const uint4* gk = (const uint4*)(kg + ((size_t)bh * 2048 + kt0 + row) * 64 + hc);
            uint4* dk = (uint4*)(&sK[row * 64 + hc]);
            dk[0] = gk[0]; dk[1] = gk[1]; dk[2] = gk[2]; dk[3] = gk[3];
            const uint4* gv = (const uint4*)(vtg + ((size_t)bh * 64 + row) * 2048 + kt0 + hc);
            uint4* dv = (uint4*)(&sVT[row * 64 + hc]);
            dv[0] = gv[0]; dv[1] = gv[1]; dv[2] = gv[2]; dv[3] = gv[3];
        }
        __syncthreads();

        // S = Q * K^T, one 16x16 frag at a time -> scaled/masked f32 to LDS
        const bool diag = (j == qt);
        {
            const ABFrag qa0 = load_frag16(sQ, wave * 16, 64, 0,  lane);
            const ABFrag qa1 = load_frag16(sQ, wave * 16, 64, 32, lane);
            for (int nf = 0; nf < 4; ++nf) {
                CFrag s;
                for (int r = 0; r < 8; ++r) s.f[r] = 0.0f;
                ABFrag kb0 = load_frag16(sK, nf * 16, 64, 0,  lane);
                ABFrag kb1 = load_frag16(sK, nf * 16, 64, 32, lane);
                s.v = __builtin_amdgcn_wmma_f32_16x16x32_bf16(
                    false, qa0.v, false, kb0.v, (short)0, s.v, false, false);
                s.v = __builtin_amdgcn_wmma_f32_16x16x32_bf16(
                    false, qa1.v, false, kb1.v, (short)0, s.v, false, false);
                const int kj = kt0 + nf * 16 + l15;
                for (int r = 0; r < 8; ++r) {
                    const int ml = r + 8 * hl;
                    float v = s.f[r] * 0.125f;            // 1/sqrt(64)
                    if (diag && kj > (q0 + wave * 16 + ml)) v = -1e30f;
                    myS[ml * 64 + nf * 16 + l15] = v;
                }
            }
        }

        // online softmax per row (row M spans one 16-lane half-wave)
        for (int r = 0; r < 8; ++r) {
            const int ml = r + 8 * hl;
            float v0 = myS[ml * 64 +  0 + l15];
            float v1 = myS[ml * 64 + 16 + l15];
            float v2 = myS[ml * 64 + 32 + l15];
            float v3 = myS[ml * 64 + 48 + l15];
            float rm = fmaxf(fmaxf(v0, v1), fmaxf(v2, v3));
            for (int off = 8; off > 0; off >>= 1)
                rm = fmaxf(rm, __shfl_xor(rm, off, 16));
            const float mnew  = fmaxf(mrow[r], rm);
            const float alpha = __expf(mrow[r] - mnew);
            mrow[r] = mnew;
            float p0 = __expf(v0 - mnew), p1 = __expf(v1 - mnew);
            float p2 = __expf(v2 - mnew), p3 = __expf(v3 - mnew);
            float rs = (p0 + p1) + (p2 + p3);
            for (int off = 8; off > 0; off >>= 1)
                rs += __shfl_xor(rs, off, 16);
            lrow[r] = lrow[r] * alpha + rs;
            o[0].f[r] *= alpha; o[1].f[r] *= alpha;
            o[2].f[r] *= alpha; o[3].f[r] *= alpha;
            myP[ml * 64 +  0 + l15] = f2bf(p0);
            myP[ml * 64 + 16 + l15] = f2bf(p1);
            myP[ml * 64 + 32 + l15] = f2bf(p2);
            myP[ml * 64 + 48 + l15] = f2bf(p3);
        }

        // O += P * V   (A-frags from per-wave LDS P, B-frags from V^T)
        const ABFrag pa0 = load_frag16(myP, 0, 64, 0,  lane);
        const ABFrag pa1 = load_frag16(myP, 0, 64, 32, lane);
        for (int nf = 0; nf < 4; ++nf) {
            ABFrag vb0 = load_frag16(sVT, nf * 16, 64, 0,  lane);
            ABFrag vb1 = load_frag16(sVT, nf * 16, 64, 32, lane);
            o[nf].v = __builtin_amdgcn_wmma_f32_16x16x32_bf16(
                false, pa0.v, false, vb0.v, (short)0, o[nf].v, false, false);
            o[nf].v = __builtin_amdgcn_wmma_f32_16x16x32_bf16(
                false, pa1.v, false, vb1.v, (short)0, o[nf].v, false, false);
        }
        __syncthreads();   // all waves done with sK/sVT before next tile
    }

    // normalize + store attn output as bf16 row [b*T+t][h*64+d]
    for (int r = 0; r < 8; ++r) {
        const int tg  = q0 + wave * 16 + r + 8 * hl;
        const float inv = 1.0f / lrow[r];
        for (int nf = 0; nf < 4; ++nf)
            attn[((size_t)(b * 2048 + tg)) * 1024 + h * 64 + nf * 16 + l15] =
                f2bf(o[nf].f[r] * inv);
    }
}

// ---------------------------------------------------------------------------
// Kernel 3: output projection  out = attn * Wo + bo   (f32 out)
// ---------------------------------------------------------------------------
__global__ __launch_bounds__(512)
__attribute__((amdgpu_waves_per_eu(1)))
void gemm_out_kernel(
    const unsigned short* __restrict__ attn,
    const unsigned short* __restrict__ woT,
    const float* __restrict__ bo, float* __restrict__ out) {
    __shared__ unsigned short sA[128 * 32];
    __shared__ unsigned short sB[128 * 32];
    CFrag c[2][2];
    const int blockM = blockIdx.y * 128, blockN = blockIdx.x * 128;
    gemm_tile_128x128(attn, woT, 1024, blockM, blockN, c, sA, sB);

    const int lane  = threadIdx.x & 31;
    const int wave  = threadIdx.x >> 5;
    const int waveM = wave >> 2, waveN = wave & 3;
    const int hl    = lane >> 4, l15 = lane & 15;
    for (int mi = 0; mi < 2; ++mi) {
        const int rowBase = blockM + waveM * 32 + mi * 16 + 8 * hl;
        for (int ni = 0; ni < 2; ++ni) {
            const int col = blockN + waveN * 32 + ni * 16 + l15;
            const float bias = bo[col];
            for (int r = 0; r < 8; ++r)
                out[(size_t)(rowBase + r) * 1024 + col] = c[mi][ni].f[r] + bias;
        }
    }
}

// ---------------------------------------------------------------------------
extern "C" void kernel_launch(void* const* d_in, const int* in_sizes, int n_in,
                              void* d_out, int out_size, void* d_ws,
                              size_t ws_size, hipStream_t stream) {
    const float* x    = (const float*)d_in[0];   // [2,2048,1024]
    const float* Wqkv = (const float*)d_in[1];   // [1024,3072]
    const float* Wo   = (const float*)d_in[2];   // [1024,1024]
    const float* bo   = (const float*)d_in[3];   // [1024]
    float* out = (float*)d_out;

    // Workspace layout (bf16 / ushort units), total 48 MB
    unsigned short* ws    = (unsigned short*)d_ws;
    unsigned short* xb    = ws;                   // 4096*1024
    unsigned short* wqkvT = xb    + 4194304;      // 3072*1024
    unsigned short* woT   = wqkvT + 3145728;      // 1024*1024
    unsigned short* qb    = woT   + 1048576;      // [2,16,2048,64]
    unsigned short* kb    = qb    + 4194304;      // [2,16,2048,64]
    unsigned short* vtb   = kb    + 4194304;      // [2,16,64,2048]
    unsigned short* attnb = vtb   + 4194304;      // 4096*1024

    convert_f32_bf16_kernel<<<4096, 256, 0, stream>>>(x, xb, 1048576);
    transpose_convert_kernel<<<dim3(3072 / 32, 1024 / 32), 256, 0, stream>>>(
        Wqkv, wqkvT, 1024, 3072);
    transpose_convert_kernel<<<dim3(1024 / 32, 1024 / 32), 256, 0, stream>>>(
        Wo, woT, 1024, 1024);

    gemm_qkv_rope_kernel<<<dim3(24, 32), 512, 0, stream>>>(xb, wqkvT, qb, kb, vtb);
    flash_attn_kernel<<<dim3(32, 16, 2), 128, 0, stream>>>(qb, kb, vtb, attnb);
    gemm_out_kernel<<<dim3(8, 32), 512, 0, stream>>>(attnb, woT, bo, out);
}
